// MIPSBruteTopK_47115791237663
// MI455X (gfx1250) — matrix operand order repaired and checked by reading, not compiled, not run on hardware
//
#include <hip/hip_runtime.h>
#include <hip/hip_bf16.h>

typedef __attribute__((ext_vector_type(16))) __bf16       v16bf;
typedef __attribute__((ext_vector_type(8)))  float        v8f;
typedef __attribute__((ext_vector_type(8)))  unsigned int v8u;

#define WROWS 128
#define LDST  132   // 128 + 4 pad floats: avoids LDS bank conflicts

// ---------- bf16 split helpers (round-to-nearest-even, bit ops only) ----------
static __device__ __forceinline__ unsigned bfpack2(float f0, float f1, float& r0, float& r1) {
    unsigned u0 = __float_as_uint(f0);
    unsigned h0 = (u0 + 0x7FFFu + ((u0 >> 16) & 1u)) & 0xFFFF0000u;
    unsigned u1 = __float_as_uint(f1);
    unsigned h1 = (u1 + 0x7FFFu + ((u1 >> 16) & 1u)) & 0xFFFF0000u;
    r0 = f0 - __uint_as_float(h0);
    r1 = f1 - __uint_as_float(h1);
    return (h0 >> 16) | h1;
}
static __device__ __forceinline__ unsigned bfpack2_only(float f0, float f1) {
    unsigned u0 = __float_as_uint(f0);
    unsigned h0 = (u0 + 0x7FFFu + ((u0 >> 16) & 1u)) & 0xFFFF0000u;
    unsigned u1 = __float_as_uint(f1);
    unsigned h1 = (u1 + 0x7FFFu + ((u1 >> 16) & 1u)) & 0xFFFF0000u;
    return (h0 >> 16) | h1;
}
static __device__ __forceinline__ v16bf as_bf(const v8u& u) {
    return __builtin_bit_cast(v16bf, u);
}

// ---------- wave32 bitonic top-128 machinery (4 elems/lane, e = r*32 + lane) ----------
static __device__ __forceinline__ bool beats(float av, int ai, float bv, int bi) {
    return (av > bv) || (av == bv && ai < bi);   // (val desc, idx asc) strict order
}
static __device__ __forceinline__ void cswapf(float& va, int& ia, float& vb, int& ib, bool desc) {
    bool bb = beats(vb, ib, va, ia);             // element b beats element a
    if (bb == desc) { float tv = va; va = vb; vb = tv; int ti = ia; ia = ib; ib = ti; }
}
static __device__ __forceinline__ void stage_xlane(float (&v)[4], int (&ix)[4],
                                                   int j, int s, int lane) {
    bool lower = ((lane & j) == 0);
#pragma unroll
    for (int r = 0; r < 4; ++r) {
        float pv = __shfl_xor(v[r], j, 32);
        int   pi = __shfl_xor(ix[r], j, 32);
        int e = (r << 5) | lane;
        bool desc = ((e & s) == 0);
        bool take_max = (lower == desc);
        bool pb = beats(pv, pi, v[r], ix[r]);
        if (pb == take_max) { v[r] = pv; ix[r] = pi; }
    }
}
static __device__ __forceinline__ void stage_reg(float (&v)[4], int (&ix)[4],
                                                 int j, int s, int lane) {
    if ((j >> 5) == 2) {   // stride 64: pairs (0,2),(1,3)
        bool d0 = (((0  | lane) & s) == 0);
        bool d1 = (((32 | lane) & s) == 0);
        cswapf(v[0], ix[0], v[2], ix[2], d0);
        cswapf(v[1], ix[1], v[3], ix[3], d1);
    } else {               // stride 32: pairs (0,1),(2,3)
        bool d0 = (((0  | lane) & s) == 0);
        bool d2 = (((64 | lane) & s) == 0);
        cswapf(v[0], ix[0], v[1], ix[1], d0);
        cswapf(v[2], ix[2], v[3], ix[3], d2);
    }
}
static __device__ __forceinline__ void sort128_desc(float (&v)[4], int (&ix)[4], int lane) {
    for (int s = 2; s <= 128; s <<= 1)
        for (int j = s >> 1; j >= 1; j >>= 1) {
            if (j >= 32) stage_reg(v, ix, j, s, lane);
            else         stage_xlane(v, ix, j, s, lane);
        }
}
// buffer (sorted desc) <-- top-128 of (buffer ∪ group), group must be sorted desc
static __device__ __forceinline__ void merge128_desc(float (&bv)[4], int (&bi)[4],
                                                     float (&gv)[4], int (&gi)[4], int lane) {
#pragma unroll
    for (int r = 0; r < 4; ++r) {                 // grev[e] = g[127-e]
        float rv = __shfl_xor(gv[3 - r], 31, 32);
        int   ri = __shfl_xor(gi[3 - r], 31, 32);
        if (beats(rv, ri, bv[r], bi[r])) { bv[r] = rv; bi[r] = ri; }
    }
    for (int j = 64; j >= 1; j >>= 1) {           // clean bitonic -> descending (s=256 => desc)
        if (j >= 32) stage_reg(bv, bi, j, 256, lane);
        else         stage_xlane(bv, bi, j, 256, lane);
    }
}

// ============================================================================
// Kernel 1: split-bf16 WMMA GEMM (128 rows x chunk) fused with per-row
//           register-resident top-128 selection; emits sorted candidates.
// ============================================================================
__global__ __launch_bounds__(512)
void mips_gemm_chunk_topk(const float* __restrict__ q,
                          const float* __restrict__ items,
                          float* __restrict__ cand_val,
                          int*   __restrict__ cand_idx,
                          int Btot, int X, int nchunks, int nsub) {
    extern __shared__ float lds[];
    const int tid  = threadIdx.x;
    const int lane = tid & 31;
    const int w    = tid >> 5;       // wave 0..15
    const int half = lane >> 4;
    const int mn   = lane & 15;
    const int rt   = w >> 1;         // GEMM row-tile 0..7
    const int ct0  = (w & 1) * 4;    // 4 col-tiles per wave
    const int chunk = blockIdx.x;
    const int rbase = blockIdx.y * WROWS;

    // ---- hoist A (16 rows x 64 K) as split-bf16 fragments ----
    v16bf ah[2], al[2];
    {
        int grow = rbase + rt * 16 + mn;
        bool rok = grow < Btot;
        int growc = rok ? grow : 0;
#pragma unroll
        for (int t = 0; t < 2; ++t) {
            v8u hu, lu;
#pragma unroll
            for (int vg = 0; vg < 8; ++vg) {
                int kb = t * 32 + ((vg & 4) << 2) + 2 * (vg & 3) + 8 * half;
                float f0 = q[growc * 64 + kb];
                float f1 = q[growc * 64 + kb + 1];
                if (!rok) { f0 = 0.0f; f1 = 0.0f; }
                float r0, r1;
                hu[vg] = bfpack2(f0, f1, r0, r1);
                lu[vg] = bfpack2_only(r0, r1);
            }
            ah[t] = as_bf(hu); al[t] = as_bf(lu);
        }
    }

    // ---- per-wave selection buffers: 8 rows, top-128 each ----
    float bv[8][4]; int bi[8][4];
#pragma unroll
    for (int rr = 0; rr < 8; ++rr)
#pragma unroll
        for (int r = 0; r < 4; ++r) { bv[rr][r] = -3.402823466e38f; bi[rr][r] = 0; }

    for (int sc = 0; sc < nsub; ++sc) {
        const int x0 = chunk * (nsub * 128) + sc * 128;

        // ---------- GEMM: 128x128 scores into LDS ----------
#pragma unroll
        for (int ctt = 0; ctt < 4; ++ctt) {
            int ct  = ct0 + ctt;
            int col = x0 + ct * 16 + mn;
            bool cok = col < X;
            size_t colc = (size_t)(cok ? col : (X - 1));
            v8f c = {};
#pragma unroll
            for (int t = 0; t < 2; ++t) {
                v8u hu, lu;
#pragma unroll
                for (int vg = 0; vg < 8; ++vg) {
                    size_t kr = (size_t)(t * 32 + ((vg & 4) << 2) + 2 * (vg & 3) + 8 * half);
                    const float* p = items + kr * (size_t)X + colc;
                    float f0 = p[0];
                    float f1 = p[(size_t)X];
                    if (!cok) { f0 = 0.0f; f1 = 0.0f; }
                    float r0, r1;
                    hu[vg] = bfpack2(f0, f1, r0, r1);
                    lu[vg] = bfpack2_only(r0, r1);
                }
                v16bf bh = as_bf(hu), bl = as_bf(lu);
                c = __builtin_amdgcn_wmma_f32_16x16x32_bf16(false, ah[t], false, bh, (short)0, c, false, false);
                c = __builtin_amdgcn_wmma_f32_16x16x32_bf16(false, ah[t], false, bl, (short)0, c, false, false);
                c = __builtin_amdgcn_wmma_f32_16x16x32_bf16(false, al[t], false, bh, (short)0, c, false, false);
            }
#pragma unroll
            for (int r = 0; r < 8; ++r) {
                int rowl = rt * 16 + half * 8 + r;   // C layout: lanes16-31 hold M=r+8
                lds[rowl * LDST + ct * 16 + mn] = c[r];
            }
        }
        __syncthreads();

        // ---------- selection: wave w owns rows w*8..w*8+7 ----------
#pragma unroll
        for (int rr = 0; rr < 8; ++rr) {
            int rowl = w * 8 + rr;
            float gv[4]; int gi[4];
#pragma unroll
            for (int r = 0; r < 4; ++r) {
                int cl = r * 32 + lane;
                float val = lds[rowl * LDST + cl];
                val = fminf(fmaxf(val, -1.0e9f), 1.0e9f);  // nan_to_num(-BIG, BIG)
                int gc = x0 + cl;
                if (gc >= X) val = -3.0e38f;
                gv[r] = val; gi[r] = gc;
            }
            float thr = __shfl(bv[rr][3], 31, 32);          // current 128th value
            bool hit = (gv[0] > thr) || (gv[1] > thr) || (gv[2] > thr) || (gv[3] > thr);
            if (__ballot(hit)) {                             // wave-uniform skip
                sort128_desc(gv, gi, lane);
                merge128_desc(bv[rr], bi[rr], gv, gi, lane);
            }
        }
        __syncthreads();
    }

    // ---- write sorted candidates: [row][chunk][128] ----
#pragma unroll
    for (int rr = 0; rr < 8; ++rr) {
        int growg = rbase + w * 8 + rr;
        if (growg < Btot) {
            size_t base = ((size_t)growg * (size_t)nchunks + (size_t)chunk) * 128u;
#pragma unroll
            for (int r = 0; r < 4; ++r) {
                cand_val[base + r * 32 + lane] = bv[rr][r];
                cand_idx[base + r * 32 + lane] = bi[rr][r];
            }
        }
    }
}

// ============================================================================
// Kernel 2: per-row merge of nchunks sorted top-128 lists -> final sorted top-k
// ============================================================================
__global__ __launch_bounds__(256)
void topk_final_merge(const float* __restrict__ cand_val,
                      const int*   __restrict__ cand_idx,
                      const long long* __restrict__ item_ids,
                      float* __restrict__ out,
                      int Btot, int X, int nchunks, int k) {
    const int lane = threadIdx.x & 31;
    const int w    = threadIdx.x >> 5;
    const int row  = blockIdx.x * 8 + w;
    if (row >= Btot) return;

    float bvv[4]; int bii[4];
#pragma unroll
    for (int r = 0; r < 4; ++r) { bvv[r] = -3.402823466e38f; bii[r] = 0; }

    size_t base = (size_t)row * (size_t)nchunks * 128u;
    for (int c = 0; c < nchunks; ++c) {
        float gv[4]; int gi[4];
#pragma unroll
        for (int r = 0; r < 4; ++r) {
            gv[r] = cand_val[base + (size_t)c * 128 + r * 32 + lane];
            gi[r] = cand_idx[base + (size_t)c * 128 + r * 32 + lane];
        }
        float thr  = __shfl(bvv[3], 31, 32);
        float gmax = __shfl(gv[0], 0, 32);   // group already sorted desc
        if (gmax > thr || c == 0)
            merge128_desc(bvv, bii, gv, gi, lane);
    }
#pragma unroll
    for (int r = 0; r < 4; ++r) {
        int e = r * 32 + lane;
        if (e < k) {
            out[(size_t)row * k + e] = bvv[r];
            int ii = bii[r];
            if (ii > X - 1) ii = X - 1;
            if (ii < 0) ii = 0;
            out[(size_t)Btot * k + (size_t)row * k + e] = (float)item_ids[ii];
        }
    }
}

// ============================================================================
extern "C" void kernel_launch(void* const* d_in, const int* in_sizes, int n_in,
                              void* d_out, int out_size, void* d_ws, size_t ws_size,
                              hipStream_t stream) {
    const float*     q     = (const float*)d_in[0];       // [B,64]
    const float*     items = (const float*)d_in[1];       // [64,X]
    const long long* ids   = (const long long*)d_in[2];   // [1,X] int64

    int Btot = in_sizes[0] / 64;
    int X    = in_sizes[1] / 64;
    int k = (Btot > 0) ? out_size / (2 * Btot) : 100;
    if (k <= 0 || k > 128) k = 100;

    // chunk count bounded by workspace: per chunk we store B*128*(4+4) bytes
    size_t per_chunk = (size_t)Btot * 128u * 8u;
    int max_chunks = per_chunk ? (int)(ws_size / per_chunk) : 1;
    if (max_chunks < 1) max_chunks = 1;
    int target = (X + 16383) / 16384;            // ~16K columns per chunk
    int nchunks = target < max_chunks ? target : max_chunks;
    if (nchunks < 1) nchunks = 1;
    int nsub = (X + nchunks * 128 - 1) / (nchunks * 128);
    nchunks = (X + nsub * 128 - 1) / (nsub * 128);

    float* cand_val = (float*)d_ws;
    int*   cand_idx = (int*)((char*)d_ws + (size_t)Btot * (size_t)nchunks * 128u * sizeof(float));

    dim3 g1(nchunks, (Btot + WROWS - 1) / WROWS);
    size_t ldsb = (size_t)WROWS * LDST * sizeof(float);
    mips_gemm_chunk_topk<<<g1, 512, ldsb, stream>>>(q, items, cand_val, cand_idx,
                                                    Btot, X, nchunks, nsub);
    topk_final_merge<<<(Btot + 7) / 8, 256, 0, stream>>>(cand_val, cand_idx, ids,
                                                         (float*)d_out, Btot, X, nchunks, k);
}
